// CustomLSTMClassifier_29703993819208
// MI455X (gfx1250) — matrix operand order repaired and checked by reading, compiled, not run on hardware
//
#include <hip/hip_runtime.h>
#include <hip/hip_bf16.h>

#define INPUT_SZ  512
#define HIDDEN    1024
#define COMB      1536      // INPUT_SZ + HIDDEN
#define BATCH     64
#define SEQ_LEN   512
#define OUT_SZ    10

typedef __bf16 bf16_t;
typedef __attribute__((ext_vector_type(8)))  bf16_t v8bf;
typedef __attribute__((ext_vector_type(16))) bf16_t v16bf;
typedef __attribute__((ext_vector_type(8)))  float  v8f;
typedef __attribute__((ext_vector_type(4)))  float  v4f;

__device__ __forceinline__ float sigmoidf_(float x) { return 1.0f / (1.0f + __expf(-x)); }

// A fragment: 16 contiguous K at half*16 of one row.
__device__ __forceinline__ v16bf load_frag_a(const bf16_t* __restrict__ row, int half) {
    v8bf lo = *(const v8bf*)(row + half * 16);
    v8bf hi = *(const v8bf*)(row + half * 16 + 8);
    return __builtin_shufflevector(lo, hi, 0, 1, 2, 3, 4, 5, 6, 7,
                                          8, 9, 10, 11, 12, 13, 14, 15);
}

// B fragment: 16 contiguous K at half*16 of the (transposed-weight) row.
__device__ __forceinline__ v16bf load_frag_b(const bf16_t* __restrict__ row, int half) {
    const bf16_t* p = row + half * 16;
    v8bf lo = *(const v8bf*)(p);
    v8bf hi = *(const v8bf*)(p + 8);
    return __builtin_shufflevector(lo, hi, 0, 1, 2, 3, 4, 5, 6, 7,
                                          8, 9, 10, 11, 12, 13, 14, 15);
}

__device__ __forceinline__ v8f wmma_bf16(v16bf a, v16bf b, v8f c) {
    return __builtin_amdgcn_wmma_f32_16x16x32_bf16(false, a, false, b,
                                                   (short)0, c, false, false);
}

// ---------------------------------------------------------------------------
// Generic one-time fp32 -> bf16 conversion (vectorized x4; n % 4 == 0 always).
// ---------------------------------------------------------------------------
__global__ void cvt_f32_to_bf16(const float* __restrict__ src,
                                bf16_t* __restrict__ dst, long long n4) {
    for (long long i = (long long)blockIdx.x * blockDim.x + threadIdx.x; i < n4;
         i += (long long)gridDim.x * blockDim.x) {
        v4f x = *(const v4f*)(src + 4 * i);
        bf16_t* d = dst + 4 * i;
        d[0] = (bf16_t)x[0]; d[1] = (bf16_t)x[1];
        d[2] = (bf16_t)x[2]; d[3] = (bf16_t)x[3];
    }
}

// Zero h (both bf16 buffers) and c (fp32).
__global__ void lstm_init_state(bf16_t* __restrict__ h0, bf16_t* __restrict__ h1,
                                float* __restrict__ c) {
    int i = blockIdx.x * blockDim.x + threadIdx.x;
    if (i < BATCH * HIDDEN) {
        h0[i] = (bf16_t)0.0f;
        h1[i] = (bf16_t)0.0f;
        c[i]  = 0.0f;
    }
}

// ---------------------------------------------------------------------------
// One fused LSTM step. Grid: (BATCH/16, HIDDEN/32), block: 128 threads.
// Wave g computes gate g for a 16(M) x 32(N) tile: two f32 accumulators,
// A fragment reused across both N sub-tiles. Bias pre-folded into acc init.
// K split into two branch-free loops (x region bf16, h region bf16).
// ---------------------------------------------------------------------------
__launch_bounds__(128)
__global__ void lstm_step(const bf16_t* __restrict__ x_bf,    // [B, S, I] bf16
                          int t,
                          const bf16_t* __restrict__ Wt,      // [4H, COMB] bf16
                          const float*  __restrict__ bF,
                          const float*  __restrict__ bI,
                          const float*  __restrict__ bO,
                          const float*  __restrict__ bC,
                          const bf16_t* __restrict__ h_in,    // [B, H] bf16
                          float*        __restrict__ c_state, // [B, H] fp32 (in place)
                          bf16_t*       __restrict__ h_out_bf,
                          float*        __restrict__ h_out_f32) {
    __shared__ float lds_g[4][16][32];

    const int im   = blockIdx.x;          // M tile (batch / 16)
    const int jn   = blockIdx.y;          // N tile pair (HIDDEN / 32)
    const int wave = threadIdx.x >> 5;    // gate id 0..3
    const int lane = threadIdx.x & 31;
    const int mrow = lane & 15;
    const int half = lane >> 4;

    // Per-gate bias pointer (select via cndmask; uniform within each wave).
    const float* bptr = bF;
    if (wave == 1) bptr = bI;
    if (wave == 2) bptr = bO;
    if (wave == 3) bptr = bC;

    // Fold bias into accumulator init: bias depends only on N = mrow.
    const float b0 = bptr[jn * 32 + mrow];
    const float b1 = bptr[jn * 32 + 16 + mrow];
    v8f acc0, acc1;
#pragma unroll
    for (int r = 0; r < 8; ++r) { acc0[r] = b0; acc1[r] = b1; }

    // B rows (weight columns) for the two N sub-tiles of this gate.
    const bf16_t* B0 = Wt + (size_t)(wave * HIDDEN + jn * 32 + mrow) * COMB;
    const bf16_t* B1 = B0 + (size_t)16 * COMB;

    // A rows for this lane's M index.
    const int m_glob = im * 16 + mrow;
    const bf16_t* Xrow = x_bf + ((size_t)m_glob * SEQ_LEN + (size_t)t) * INPUT_SZ;
    const bf16_t* Hrow = h_in + (size_t)m_glob * HIDDEN;

#pragma unroll 4
    for (int kb = 0; kb < INPUT_SZ; kb += 32) {
        v16bf a = load_frag_a(Xrow + kb, half);
        acc0 = wmma_bf16(a, load_frag_b(B0 + kb, half), acc0);
        acc1 = wmma_bf16(a, load_frag_b(B1 + kb, half), acc1);
    }
#pragma unroll 4
    for (int kb = 0; kb < HIDDEN; kb += 32) {
        v16bf a = load_frag_a(Hrow + kb, half);
        acc0 = wmma_bf16(a, load_frag_b(B0 + INPUT_SZ + kb, half), acc0);
        acc1 = wmma_bf16(a, load_frag_b(B1 + INPUT_SZ + kb, half), acc1);
    }

    // Spill gate tiles to LDS: VGPR r -> M = half*8 + r, N = mrow (+16).
#pragma unroll
    for (int r = 0; r < 8; ++r) {
        lds_g[wave][half * 8 + r][mrow]      = acc0[r];
        lds_g[wave][half * 8 + r][16 + mrow] = acc1[r];
    }
    __syncthreads();

    // Elementwise LSTM update: 16 x 32 = 512 elements over 128 threads.
    for (int e = threadIdx.x; e < 512; e += 128) {
        const int ml = e >> 5, nl = e & 31;
        const int ng = jn * 32 + nl;
        const int mg = im * 16 + ml;
        const float f  = sigmoidf_(lds_g[0][ml][nl]);
        const float ii = sigmoidf_(lds_g[1][ml][nl]);
        const float oo = sigmoidf_(lds_g[2][ml][nl]);
        const float gg = tanhf(lds_g[3][ml][nl]);
        const size_t idx = (size_t)mg * HIDDEN + ng;
        const float cn = f * c_state[idx] + ii * gg;
        c_state[idx] = cn;
        const float hn = oo * tanhf(cn);
        h_out_bf[idx]  = (bf16_t)hn;
        h_out_f32[idx] = hn;
    }
}

// Final classifier: logits[b][o] = h[b] . out_w[o] + out_b[o]  (64x10 outputs)
__global__ void lstm_logits(const float* __restrict__ h,
                            const float* __restrict__ out_w,
                            const float* __restrict__ out_b,
                            float* __restrict__ out) {
    const int o = blockIdx.x;
    const int b = threadIdx.x;
    if (o < OUT_SZ && b < BATCH) {
        const float* hr = h + (size_t)b * HIDDEN;
        const float* wr = out_w + (size_t)o * HIDDEN;
        float s = out_b[o];
        for (int k = 0; k < HIDDEN; ++k) s += hr[k] * wr[k];
        out[b * OUT_SZ + o] = s;
    }
}

extern "C" void kernel_launch(void* const* d_in, const int* in_sizes, int n_in,
                              void* d_out, int out_size, void* d_ws, size_t ws_size,
                              hipStream_t stream) {
    const float* x    = (const float*)d_in[0];   // input_seq [64,512,512]
    const float* Wf_w = (const float*)d_in[1];
    const float* Wf_b = (const float*)d_in[2];
    const float* Wi_w = (const float*)d_in[3];
    const float* Wi_b = (const float*)d_in[4];
    const float* Wo_w = (const float*)d_in[5];
    const float* Wo_b = (const float*)d_in[6];
    const float* Wc_w = (const float*)d_in[7];
    const float* Wc_b = (const float*)d_in[8];
    const float* ow   = (const float*)d_in[9];
    const float* ob   = (const float*)d_in[10];

    const size_t perW = (size_t)HIDDEN * COMB;        // one gate weight matrix
    const size_t nX   = (size_t)BATCH * SEQ_LEN * INPUT_SZ;

    // Workspace layout (~47 MB total)
    char* ws = (char*)d_ws;
    bf16_t* Wt = (bf16_t*)ws;                         // [4H, COMB] bf16: 12.6 MB
    size_t off = 4 * perW * sizeof(bf16_t);
    bf16_t* x_bf = (bf16_t*)(ws + off); off += nX * sizeof(bf16_t);   // 33.6 MB
    bf16_t* hbuf0 = (bf16_t*)(ws + off); off += (size_t)BATCH * HIDDEN * sizeof(bf16_t);
    bf16_t* hbuf1 = (bf16_t*)(ws + off); off += (size_t)BATCH * HIDDEN * sizeof(bf16_t);
    float*  c_st  = (float*)(ws + off);  off += (size_t)BATCH * HIDDEN * sizeof(float);
    float*  h_f32 = (float*)(ws + off);

    // One-time bf16 packing: 4 gate weight matrices + full input sequence.
    cvt_f32_to_bf16<<<1024, 256, 0, stream>>>(Wf_w, Wt + 0 * perW, (long long)(perW / 4));
    cvt_f32_to_bf16<<<1024, 256, 0, stream>>>(Wi_w, Wt + 1 * perW, (long long)(perW / 4));
    cvt_f32_to_bf16<<<1024, 256, 0, stream>>>(Wo_w, Wt + 2 * perW, (long long)(perW / 4));
    cvt_f32_to_bf16<<<1024, 256, 0, stream>>>(Wc_w, Wt + 3 * perW, (long long)(perW / 4));
    cvt_f32_to_bf16<<<2048, 256, 0, stream>>>(x,    x_bf,          (long long)(nX / 4));
    lstm_init_state<<<(BATCH * HIDDEN + 255) / 256, 256, 0, stream>>>(hbuf0, hbuf1, c_st);

    bf16_t* hin = hbuf0;
    bf16_t* hout = hbuf1;
    for (int t = 0; t < SEQ_LEN; ++t) {
        lstm_step<<<dim3(BATCH / 16, HIDDEN / 32), 128, 0, stream>>>(
            x_bf, t, Wt, Wf_b, Wi_b, Wo_b, Wc_b, hin, c_st, hout, h_f32);
        bf16_t* tmp = hin; hin = hout; hout = tmp;
    }

    lstm_logits<<<OUT_SZ, 64, 0, stream>>>(h_f32, ow, ob, (float*)d_out);
}
